// PtrNet1_80015240724642
// MI455X (gfx1250) — matrix operand order repaired and checked by reading, not compiled, not run on hardware
//
#include <hip/hip_runtime.h>
#include <stdint.h>

// Pointer-network greedy decode for MI455X (gfx1250, wave32, WMMA).
// B=512, N=200, E=H=128. Strategy:
//  - precompute emb (f16) and uref1/uref2 (f32, L2-resident: 105MB < 192MB L2)
//    with v_wmma_f32_16x16x32_f16 GEMMs
//  - persistent decode kernel: 32 WGs x 512 threads (16 waves), each WG owns a
//    16-row batch tile. WMMA GEMMs on waves 0-7 (LSTM cell state lives in
//    registers for all 200 steps); attention/tanh (the VALU-bound part) runs
//    on all 16 waves.

#define B_SZ  512
#define N_SZ  200
#define E_SZ  128
#define H_SZ  128
#define H4_SZ 512
#define DEC_T 512  // decode block size (16 waves)

typedef _Float16 h16;
typedef h16   v16h __attribute__((ext_vector_type(16)));
typedef float v8f  __attribute__((ext_vector_type(8)));

union FragU { uint4 q[2]; v16h v; };

__device__ __forceinline__ float fast_tanh(float x) {
  float e = __expf(2.0f * x);
  return 1.0f - 2.0f / (e + 1.0f);
}
__device__ __forceinline__ float fast_sigm(float x) {
  return 1.0f / (1.0f + __expf(-x));
}
__device__ __forceinline__ v8f zero8() {
  v8f z;
#pragma unroll
  for (int i = 0; i < 8; ++i) z[i] = 0.0f;
  return z;
}

// A-matrix fragment (16x32 f16, M x K) per ISA layout:
// lanes 0-15: row M=lane, K = kc*32 + {0..7, 16..23}
// lanes 16-31: row M=lane-16, K = kc*32 + {8..15, 24..31}
__device__ __forceinline__ v16h load_frag_A(const h16* base, int ldk, int kc, int lane) {
  int hf  = lane >> 4;
  int row = lane & 15;
  const h16* p = base + row * ldk + kc * 32 + hf * 8;
  FragU u;
  u.q[0] = *(const uint4*)(p);
  u.q[1] = *(const uint4*)(p + 16);
  return u.v;
}

// B-matrix fragment (32x16 f16, K x N) for out = A @ W^T, W row-major (OUT,K):
// lanes 0-15: col n0+lane, K = kc*32 + {0..7, 16..23}
// lanes 16-31: col n0+lane-16, K = kc*32 + {8..15, 24..31}
__device__ __forceinline__ v16h load_frag_B(const h16* W, int ldk, int n0, int kc, int lane) {
  int hf  = lane >> 4;
  int col = n0 + (lane & 15);
  const h16* p = W + col * ldk + kc * 32 + hf * 8;
  FragU u;
  u.q[0] = *(const uint4*)(p);
  u.q[1] = *(const uint4*)(p + 16);
  return u.v;
}

__device__ __forceinline__ v8f wmma16(v16h a, v16h b, v8f c) {
  return __builtin_amdgcn_wmma_f32_16x16x32_f16(false, a, false, b, (short)0, c,
                                                false, false);
}

// ---------------------------------------------------------------------------
__global__ void cvt_f16_kernel(const float* __restrict__ src,
                               h16* __restrict__ dst, int n) {
  int i = blockIdx.x * blockDim.x + threadIdx.x;
  if (i < n) dst[i] = (h16)src[i];
}

__global__ void emb_kernel(const float* __restrict__ x,
                           const float* __restrict__ W_emb,
                           h16* __restrict__ emb) {
  int idx = blockIdx.x * blockDim.x + threadIdx.x;
  if (idx >= B_SZ * N_SZ * E_SZ) return;
  int bn = idx >> 7, e = idx & 127;
  const float* xp = x + bn * 3;
  const float* w  = W_emb + e * 3;
  emb[idx] = (h16)(xp[0] * w[0] + xp[1] * w[1] + xp[2] * w[2]);
}

// uref{1,2} = emb @ Wref{1,2}^T + bref{1,2}; M=B*N, N=128, K=128 (f32 out)
__global__ __launch_bounds__(256) void uref_gemm_kernel(
    const h16* __restrict__ emb,
    const h16* __restrict__ Wr1, const float* __restrict__ br1,
    const h16* __restrict__ Wr2, const float* __restrict__ br2,
    float* __restrict__ u1, float* __restrict__ u2) {
  __shared__ __attribute__((aligned(16))) h16 sA[16 * 128];
  int m0 = blockIdx.x * 16;
  ((uint4*)sA)[threadIdx.x] =
      ((const uint4*)(emb + (size_t)m0 * 128))[threadIdx.x];
  __syncthreads();
  int wv = threadIdx.x >> 5, lane = threadIdx.x & 31;
  int hf = lane >> 4, nn = lane & 15;
  int n0 = wv * 16;
  v16h af[4];
#pragma unroll
  for (int kc = 0; kc < 4; ++kc) af[kc] = load_frag_A(sA, 128, kc, lane);
  v8f acc1 = zero8(), acc2 = zero8();
#pragma unroll
  for (int kc = 0; kc < 4; ++kc) {
    acc1 = wmma16(af[kc], load_frag_B(Wr1, 128, n0, kc, lane), acc1);
    acc2 = wmma16(af[kc], load_frag_B(Wr2, 128, n0, kc, lane), acc2);
  }
  int col = n0 + nn;
  float b1 = br1[col], b2 = br2[col];
#pragma unroll
  for (int r = 0; r < 8; ++r) {
    int mrow = r + 8 * hf;
    u1[(size_t)(m0 + mrow) * 128 + col] = acc1[r] + b1;
    u2[(size_t)(m0 + mrow) * 128 + col] = acc2[r] + b2;
  }
}

// ---------------------------------------------------------------------------
// Persistent decode: 32 blocks x 512 threads; block b owns batch rows
// [16b, 16b+16). Runs the full 200-step greedy decode.
__global__ __launch_bounds__(DEC_T) void decode_kernel(
    const float* __restrict__ uref1, const float* __restrict__ uref2,
    const h16* __restrict__ embq,
    const h16* __restrict__ Wih_h, const h16* __restrict__ Whh_h,
    const h16* __restrict__ Wq_h,  const h16* __restrict__ Wq2_h,
    const float* __restrict__ bq,  const float* __restrict__ bq2,
    const float* __restrict__ vec1, const float* __restrict__ vec2,
    const float* __restrict__ bih, const float* __restrict__ bhh,
    const float* __restrict__ dec0,
    float* __restrict__ out) {
  __shared__ __attribute__((aligned(16))) h16   sA[16 * 256];   // [dec | h]
  __shared__ __attribute__((aligned(16))) float sU[16 * 128];   // u1 / u1p
  __shared__ __attribute__((aligned(16))) float sS[16 * N_SZ];  // scores/probs
  __shared__ __attribute__((aligned(16))) float sBias[512];     // bih + bhh
  __shared__ __attribute__((aligned(16))) float sVec1[128];
  __shared__ __attribute__((aligned(16))) float sVec2[128];
  __shared__ float    sRed[DEC_T];
  __shared__ int      sRedI[DEC_T];
  __shared__ unsigned sMask[16 * 7];  // 200-bit mask per row
  __shared__ float    sScal[16];      // per-row scalar (inv-sum / max)
  __shared__ int      sNxt[16];
  __shared__ float    sLL[16];

  const int tid  = threadIdx.x;
  const int wv   = tid >> 5;
  const int lane = tid & 31;
  const int hf   = lane >> 4;
  const int nn   = lane & 15;
  const int bg0  = blockIdx.x * 16;

  // ---- init
  sBias[tid] = bih[tid] + bhh[tid];
  if (tid < 128) { sVec1[tid] = vec1[tid]; sVec2[tid] = vec2[tid]; }
  if (tid < 16 * 7) sMask[tid] = 0u;
  if (tid < 16) sLL[tid] = 0.0f;
  for (int idx = tid; idx < 16 * 128; idx += DEC_T) {
    int b = idx >> 7, hh = idx & 127;
    sA[b * 256 + hh]       = (h16)dec0[hh];
    sA[b * 256 + 128 + hh] = (h16)0.0f;   // h = 0
  }
  float cReg[8];
#pragma unroll
  for (int r = 0; r < 8; ++r) cReg[r] = 0.0f;  // c = 0 (registers, waves 0-7)
  __syncthreads();

  for (int step = 0; step < N_SZ; ++step) {
    // ---- LSTM gates: (16x256)[dec|h] @ [Wih|Whh]^T -> (16x512), waves 0-7.
    // wave wv owns columns {16wv, 16wv+128, 16wv+256, 16wv+384} = i,f,g,o
    v8f accG[4];
    if (wv < 8) {
#pragma unroll
      for (int g = 0; g < 4; ++g) accG[g] = zero8();
#pragma unroll
      for (int kc = 0; kc < 8; ++kc) {
        v16h a = load_frag_A(sA, 256, kc, lane);
        const h16* W = (kc < 4) ? Wih_h : Whh_h;
        int kk = kc & 3;
#pragma unroll
        for (int g = 0; g < 4; ++g) {
          v16h b = load_frag_B(W, 128, wv * 16 + 128 * g, kk, lane);
          accG[g] = wmma16(a, b, accG[g]);
        }
      }
    }
    __syncthreads();  // gate reads of sA (h part) done before overwrite

    // ---- LSTM elementwise update (c stays in registers), write h into sA
    if (wv < 8) {
      int n = wv * 16 + nn;
#pragma unroll
      for (int r = 0; r < 8; ++r) {
        int mrow = r + 8 * hf;
        float gi = accG[0][r] + sBias[n];
        float gf = accG[1][r] + sBias[128 + n];
        float gg = accG[2][r] + sBias[256 + n];
        float go = accG[3][r] + sBias[384 + n];
        float cN = fast_sigm(gf) * cReg[r] + fast_sigm(gi) * fast_tanh(gg);
        cReg[r]  = cN;
        float hN = fast_sigm(go) * fast_tanh(cN);
        sA[mrow * 256 + 128 + n] = (h16)hN;
      }
    }
    __syncthreads();

    // ---- u1 = h @ Wq^T + bq  (wave wv -> output cols [16wv,16wv+16))
    if (wv < 8) {
      v8f acc = zero8();
#pragma unroll
      for (int kc = 0; kc < 4; ++kc) {
        v16h a = load_frag_A(sA + 128, 256, kc, lane);
        v16h b = load_frag_B(Wq_h, 128, wv * 16, kc, lane);
        acc = wmma16(a, b, acc);
      }
      int col = wv * 16 + nn;
      float bb = bq[col];
#pragma unroll
      for (int r = 0; r < 8; ++r) sU[(r + 8 * hf) * 128 + col] = acc[r] + bb;
    }
    __syncthreads();

    // ---- glimpse softmax weights, fused exp:
    // a[b,n] = exp(vec1 . tanh(u1[b]+uref1[b,n]))  (masked -> exactly 0)
    // scores are bounded by |vec1|_1 ~ 10.2 so no max-subtraction needed.
    for (int p = tid; p < 16 * N_SZ; p += DEC_T) {
      int b = p / N_SZ, n = p - b * N_SZ;
      unsigned mbit = (sMask[b * 7 + (n >> 5)] >> (n & 31)) & 1u;
      const float4* ur =
          (const float4*)(uref1 + ((size_t)(bg0 + b) * N_SZ + n) * 128);
      const float4* uu = (const float4*)(sU + b * 128);
      const float4* vv = (const float4*)sVec1;
      float acc = 0.0f;
#pragma unroll 4
      for (int q4 = 0; q4 < 32; ++q4) {
        float4 a = ur[q4], c = uu[q4], v = vv[q4];
        acc += v.x * fast_tanh(c.x + a.x);
        acc += v.y * fast_tanh(c.y + a.y);
        acc += v.z * fast_tanh(c.z + a.z);
        acc += v.w * fast_tanh(c.w + a.w);
      }
      sS[p] = mbit ? 0.0f : __expf(acc);
    }
    __syncthreads();

    // ---- row sums -> 1/sum (32 threads per row)
    {
      int b = tid >> 5, sub = tid & 31;
      float s = 0.0f;
      for (int n = sub; n < N_SZ; n += 32) s += sS[b * N_SZ + n];
      sRed[tid] = s;
      __syncthreads();
      if (sub == 0) {
        float t = 0.0f;
#pragma unroll
        for (int k = 0; k < 32; ++k) t += sRed[(b << 5) + k];
        sScal[b] = 1.0f / t;
      }
    }
    __syncthreads();

    // ---- q = a @ uref1 ; stage f16 into sA[:,0:128]
    for (int idx = tid; idx < 2048; idx += DEC_T) {
      int b = idx >> 7, hh = idx & 127;
      const float* ur = uref1 + (size_t)(bg0 + b) * N_SZ * 128 + hh;
      float acc = 0.0f;
      for (int n = 0; n < N_SZ; ++n) acc += sS[b * N_SZ + n] * ur[(size_t)n * 128];
      sA[b * 256 + hh] = (h16)(acc * sScal[b]);
    }
    __syncthreads();

    // ---- u1p = q @ Wq2^T + bq2
    if (wv < 8) {
      v8f acc = zero8();
#pragma unroll
      for (int kc = 0; kc < 4; ++kc) {
        v16h a = load_frag_A(sA, 256, kc, lane);
        v16h b = load_frag_B(Wq2_h, 128, wv * 16, kc, lane);
        acc = wmma16(a, b, acc);
      }
      int col = wv * 16 + nn;
      float bb = bq2[col];
#pragma unroll
      for (int r = 0; r < 8; ++r) sU[(r + 8 * hf) * 128 + col] = acc[r] + bb;
    }
    __syncthreads();

    // ---- pointer logits = vec2 . (10 * tanh(u1p + uref2)) (masked -> -1e8)
    for (int p = tid; p < 16 * N_SZ; p += DEC_T) {
      int b = p / N_SZ, n = p - b * N_SZ;
      unsigned mbit = (sMask[b * 7 + (n >> 5)] >> (n & 31)) & 1u;
      const float4* ur =
          (const float4*)(uref2 + ((size_t)(bg0 + b) * N_SZ + n) * 128);
      const float4* uu = (const float4*)(sU + b * 128);
      const float4* vv = (const float4*)sVec2;
      float acc = 0.0f;
#pragma unroll 4
      for (int q4 = 0; q4 < 32; ++q4) {
        float4 a = ur[q4], c = uu[q4], v = vv[q4];
        acc += v.x * fast_tanh(c.x + a.x);
        acc += v.y * fast_tanh(c.y + a.y);
        acc += v.z * fast_tanh(c.z + a.z);
        acc += v.w * fast_tanh(c.w + a.w);
      }
      sS[p] = mbit ? -1.0e8f : (10.0f * acc);
    }
    __syncthreads();

    // ---- argmax + log-softmax ll accumulation (32 threads per row)
    {
      int b = tid >> 5, sub = tid & 31;
      float m = -3.0e38f; int mi = 0;
      for (int n = sub; n < N_SZ; n += 32) {
        float v = sS[b * N_SZ + n];
        if (v > m) { m = v; mi = n; }
      }
      sRed[tid] = m; sRedI[tid] = mi;
      __syncthreads();
      if (sub == 0) {
        float bm = -3.0e38f; int bi = 0;
#pragma unroll
        for (int k = 0; k < 32; ++k) {
          float v = sRed[(b << 5) + k]; int ii = sRedI[(b << 5) + k];
          if (v > bm || (v == bm && ii < bi)) { bm = v; bi = ii; }
        }
        sNxt[b] = bi; sScal[b] = bm;
      }
      __syncthreads();
      float e = 0.0f;
      for (int n = sub; n < N_SZ; n += 32)
        e += __expf(sS[b * N_SZ + n] - sScal[b]);
      sRed[tid] = e;
      __syncthreads();
      if (sub == 0) {
        float t = 0.0f;
#pragma unroll
        for (int k = 0; k < 32; ++k) t += sRed[(b << 5) + k];
        sLL[b] -= __logf(t);  // chosen log_p = -log(sum(exp(l - max)))
        int nx = sNxt[b];
        out[(size_t)(bg0 + b) * N_SZ + step] = (float)nx;  // pi
        sMask[b * 7 + (nx >> 5)] |= (1u << (nx & 31));
      }
    }
    __syncthreads();

    // ---- gather next decoder input: dec = emb[b, nxt, :]
    for (int idx = tid; idx < 2048; idx += DEC_T) {
      int b = idx >> 7, hh = idx & 127;
      sA[b * 256 + hh] =
          embq[((size_t)(bg0 + b) * N_SZ + sNxt[b]) * 128 + hh];
    }
    __syncthreads();
  }

  if (tid < 16) out[(size_t)B_SZ * N_SZ + bg0 + tid] = sLL[tid];
}

// ---------------------------------------------------------------------------
extern "C" void kernel_launch(void* const* d_in, const int* in_sizes, int n_in,
                              void* d_out, int out_size, void* d_ws,
                              size_t ws_size, hipStream_t stream) {
  (void)in_sizes; (void)n_in; (void)out_size; (void)ws_size;
  const float* x     = (const float*)d_in[0];
  const float* W_emb = (const float*)d_in[1];
  const float* dec0  = (const float*)d_in[2];
  const float* vec1  = (const float*)d_in[3];
  const float* vec2  = (const float*)d_in[4];
  const float* Wq    = (const float*)d_in[5];
  const float* bq    = (const float*)d_in[6];
  const float* Wref  = (const float*)d_in[7];
  const float* bref  = (const float*)d_in[8];
  const float* Wq2   = (const float*)d_in[9];
  const float* bq2   = (const float*)d_in[10];
  const float* Wref2 = (const float*)d_in[11];
  const float* bref2 = (const float*)d_in[12];
  const float* Wih   = (const float*)d_in[13];
  const float* Whh   = (const float*)d_in[14];
  const float* bih   = (const float*)d_in[15];
  const float* bhh   = (const float*)d_in[16];

  char* p = (char*)d_ws;
  auto carve = [&](size_t bytes) -> char* {
    char* r = p;
    p += (bytes + 255) & ~(size_t)255;
    return r;
  };
  h16*   emb_h   = (h16*)carve((size_t)B_SZ * N_SZ * E_SZ * sizeof(h16));
  float* uref1   = (float*)carve((size_t)B_SZ * N_SZ * H_SZ * sizeof(float));
  float* uref2   = (float*)carve((size_t)B_SZ * N_SZ * H_SZ * sizeof(float));
  h16*   Wih_h   = (h16*)carve((size_t)H4_SZ * E_SZ * sizeof(h16));
  h16*   Whh_h   = (h16*)carve((size_t)H4_SZ * H_SZ * sizeof(h16));
  h16*   Wq_h    = (h16*)carve((size_t)H_SZ * H_SZ * sizeof(h16));
  h16*   Wq2_h   = (h16*)carve((size_t)H_SZ * H_SZ * sizeof(h16));
  h16*   Wref_h  = (h16*)carve((size_t)H_SZ * H_SZ * sizeof(h16));
  h16*   Wref2_h = (h16*)carve((size_t)H_SZ * H_SZ * sizeof(h16));

  // f32 -> f16 weight copies
  cvt_f16_kernel<<<(H4_SZ * E_SZ + 255) / 256, 256, 0, stream>>>(Wih, Wih_h, H4_SZ * E_SZ);
  cvt_f16_kernel<<<(H4_SZ * H_SZ + 255) / 256, 256, 0, stream>>>(Whh, Whh_h, H4_SZ * H_SZ);
  cvt_f16_kernel<<<(H_SZ * H_SZ + 255) / 256, 256, 0, stream>>>(Wq, Wq_h, H_SZ * H_SZ);
  cvt_f16_kernel<<<(H_SZ * H_SZ + 255) / 256, 256, 0, stream>>>(Wq2, Wq2_h, H_SZ * H_SZ);
  cvt_f16_kernel<<<(H_SZ * H_SZ + 255) / 256, 256, 0, stream>>>(Wref, Wref_h, H_SZ * H_SZ);
  cvt_f16_kernel<<<(H_SZ * H_SZ + 255) / 256, 256, 0, stream>>>(Wref2, Wref2_h, H_SZ * H_SZ);

  // emb (f16)
  emb_kernel<<<(B_SZ * N_SZ * E_SZ + 255) / 256, 256, 0, stream>>>(x, W_emb, emb_h);

  // uref1 / uref2 via WMMA GEMM
  uref_gemm_kernel<<<(B_SZ * N_SZ) / 16, 256, 0, stream>>>(
      emb_h, Wref_h, bref, Wref2_h, bref2, uref1, uref2);

  // persistent 200-step decode
  decode_kernel<<<B_SZ / 16, DEC_T, 0, stream>>>(
      uref1, uref2, emb_h, Wih_h, Whh_h, Wq_h, Wq2_h, bq, bq2, vec1, vec2,
      bih, bhh, dec0, (float*)d_out);
}
